// Recurrence_32349693673707
// MI455X (gfx1250) — compile-verified, matrix operand order checked by reading, not run on hardware
//
#include <hip/hip_runtime.h>
#include <hip/hip_bf16.h>

typedef __attribute__((ext_vector_type(16))) _Float16 v16h;
typedef __attribute__((ext_vector_type(8)))  _Float16 v8h;
typedef __attribute__((ext_vector_type(8)))  float    v8f;

// ---- output row layout ----
#define ROWSZ 26258
#define OFF_PROBS 1
#define OFF_V 17
#define OFF_HT 18
#define OFF_R 530
#define OFF_WR 786
#define OFF_U 1298
#define OFF_WW 1426
#define OFF_M 1554
#define OFF_P 9746
#define OFF_L 9874

__device__ __forceinline__ float sigf(float x) { return 1.f / (1.f + expf(-x)); }
__device__ __forceinline__ float softplusf(float x) { return x > 20.f ? x : log1pf(expf(x)); }

// ---------- f32 -> f16 conversion with zero tail padding ----------
__global__ void cvt_pad_f16_k(const float* __restrict__ s, _Float16* __restrict__ d,
                              int nvalid, int ntotal) {
    int i = blockIdx.x * blockDim.x + threadIdx.x;
    if (i < ntotal) d[i] = (i < nvalid) ? (_Float16)s[i] : (_Float16)0.f;
}

__global__ void pad_bias_k(const float* __restrict__ s, float* __restrict__ d,
                           int nvalid, int ntotal) {
    int i = blockIdx.x * blockDim.x + threadIdx.x;
    if (i < ntotal) d[i] = (i < nvalid) ? s[i] : 0.f;
}

// ---------- unpack rnn_hxs into state buffers (+ f16 mirrors of h) ----------
__global__ void init_state_k(const float* __restrict__ hx,
                             float* h0, float* h1, _Float16* h0h, _Float16* h1h,
                             float* r, float* wr,
                             float* u, float* ww, float* M, float* p, float* L) {
    int i = blockIdx.x * blockDim.x + threadIdx.x;
    if (i >= 128 * 26240) return;
    int n = i / 26240;
    int c = i % 26240 + 18;
    float v = hx[(size_t)n * ROWSZ + c];
    if (c < 530) {
        int l = c - 18;
        if (l < 256) { h0[n*256+l] = v; h0h[n*256+l] = (_Float16)v; }
        else         { h1[n*256+l-256] = v; h1h[n*256+l-256] = (_Float16)v; }
    }
    else if (c < 1298) { if (c < 786) r[n*256 + (c-530)] = v; else wr[n*512 + (c-786)] = v; }
    else if (c < 1426) u[n*128 + (c-1298)] = v;
    else if (c < 1554) ww[n*128 + (c-1426)] = v;
    else if (c < 9746) M[(size_t)n*8192 + (c-1554)] = v;
    else if (c < 9874) p[n*128 + (c-9746)] = v;
    else               L[(size_t)n*16384 + (c-9874)] = v;
}

// ---------- x = [obs_t | r] in f16 ----------
__global__ void build_x_k(const float* __restrict__ in_t, const float* __restrict__ r,
                          _Float16* __restrict__ xh) {
    int i = blockIdx.x * blockDim.x + threadIdx.x;     // 128*384
    if (i >= 128 * 384) return;
    int n = i / 384, c = i % 384;
    float v = (c < 128) ? in_t[n * 129 + c] : r[n * 256 + (c - 128)];
    xh[i] = (_Float16)v;
}

// ---------- WMMA GEMM: C[M x Nc] = A[M x K](f16) @ W[Nc x K](f16)^T + bias ----------
// One wave per 16x16 C tile. Fragments load as 2x b128 per lane per 32-K step
// (wave32 layout: lane kh-half holds K chunks [kh*8,kh*8+8) and [16+kh*8,...)).
// K templated -> fully unrolled, two independent WMMA accumulator chains.
template <int K>
__global__ void wmma_gemm_k(const _Float16* __restrict__ A, const _Float16* __restrict__ W,
                            const float* __restrict__ bias, float* __restrict__ C,
                            int M, int Nc, int ldc) {
    int wave = (int)((blockIdx.x * blockDim.x + threadIdx.x) >> 5);
    int lane = threadIdx.x & 31;
    int tilesn = Nc >> 4;
    int tilesm = M >> 4;
    int tm = wave / tilesn;
    int tn = wave - tm * tilesn;
    if (tm >= tilesm) return;                 // wave-uniform branch: EXEC stays all-ones
    int m0 = tm << 4, n0 = tn << 4;
    int lm = lane & 15, kh = lane >> 4;
    const _Float16* arow = A + (size_t)(m0 + lm) * K + kh * 8;
    const _Float16* wrow = W + (size_t)(n0 + lm) * K + kh * 8;
    v8f acc0 = {}, acc1 = {};
#pragma unroll
    for (int k0 = 0; k0 < K; k0 += 64) {
        v8h a0 = *(const v8h*)(arow + k0);
        v8h a1 = *(const v8h*)(arow + k0 + 16);
        v8h b0 = *(const v8h*)(wrow + k0);
        v8h b1 = *(const v8h*)(wrow + k0 + 16);
        v8h a2 = *(const v8h*)(arow + k0 + 32);
        v8h a3 = *(const v8h*)(arow + k0 + 48);
        v8h b2 = *(const v8h*)(wrow + k0 + 32);
        v8h b3 = *(const v8h*)(wrow + k0 + 48);
        v16h fa0 = __builtin_shufflevector(a0, a1, 0,1,2,3,4,5,6,7,8,9,10,11,12,13,14,15);
        v16h fb0 = __builtin_shufflevector(b0, b1, 0,1,2,3,4,5,6,7,8,9,10,11,12,13,14,15);
        acc0 = __builtin_amdgcn_wmma_f32_16x16x32_f16(false, fa0, false, fb0, (short)0, acc0,
                                                      false, false);
        v16h fa1 = __builtin_shufflevector(a2, a3, 0,1,2,3,4,5,6,7,8,9,10,11,12,13,14,15);
        v16h fb1 = __builtin_shufflevector(b2, b3, 0,1,2,3,4,5,6,7,8,9,10,11,12,13,14,15);
        acc1 = __builtin_amdgcn_wmma_f32_16x16x32_f16(false, fa1, false, fb1, (short)0, acc1,
                                                      false, false);
    }
    v8f acc = acc0 + acc1;
    int cn = n0 + lm;
    float bb = bias[cn];
    int mo = m0 + (kh << 3);
    float* crow = C + (size_t)mo * ldc + cn;
#pragma unroll
    for (int v = 0; v < 8; ++v) crow[(size_t)v * ldc] = acc[v] + bb;
}

// ---------- GRU gate combine: writes f32 and f16 h_new ----------
__global__ void gru_combine_k(const float* __restrict__ gi, const float* __restrict__ gh,
                              const float* __restrict__ h, float* __restrict__ hn,
                              _Float16* __restrict__ hnh) {
    int i = blockIdx.x * blockDim.x + threadIdx.x;     // 128*256
    if (i >= 128 * 256) return;
    int n = i / 256, c = i % 256;
    const float* gin = gi + (size_t)n * 768;
    const float* ghn = gh + (size_t)n * 768;
    float r = sigf(gin[c] + ghn[c]);
    float z = sigf(gin[256 + c] + ghn[256 + c]);
    float nn = tanhf(gin[512 + c] + r * ghn[512 + c]);
    float o = (1.f - z) * nn + z * h[i];
    hn[i] = o;
    hnh[i] = (_Float16)o;
}

// ---------- commit h0/h1 (+f16), build hT f32 + relu(hT) f16, emit hT/act ----------
__global__ void commit_h_k(const float* __restrict__ h0n, const float* __restrict__ h1n,
                           float* __restrict__ h0, float* __restrict__ h1,
                           _Float16* __restrict__ h0h, _Float16* __restrict__ h1h,
                           float* __restrict__ hT, _Float16* __restrict__ hTrh,
                           float* __restrict__ out_t, const float* __restrict__ in_t) {
    int i = blockIdx.x * blockDim.x + threadIdx.x;     // 128*512
    if (i >= 128 * 512) return;
    int n = i / 512, c = i % 512;
    float v = (c < 256) ? h0n[n * 256 + c] : h1n[n * 256 + (c - 256)];
    hT[i] = v;
    hTrh[i] = (_Float16)fmaxf(v, 0.f);
    if (c < 256) { h0[n * 256 + c] = v; h0h[n * 256 + c] = (_Float16)v; }
    else         { h1[n * 256 + (c-256)] = v; h1h[n * 256 + (c-256)] = (_Float16)v; }
    out_t[(size_t)n * ROWSZ + OFF_HT + c] = v;
    if (c == 0) out_t[(size_t)n * ROWSZ + 0] = in_t[n * 129 + 128];  // action passthrough
}

// ---------- xi split activations ----------
__global__ void xi_act_k(const float* __restrict__ xin, float* __restrict__ xout) {
    int i = blockIdx.x * blockDim.x + threadIdx.x;     // 128*471
    if (i >= 128 * 471) return;
    int n = i / 471, c = i % 471;
    const float* row = xin + (size_t)n * 480;
    float v = row[c], o;
    if (c < 256)        o = v;                 // Kr
    else if (c < 260)   o = softplusf(v);      // br
    else if (c < 324)   o = v;                 // kw
    else if (c == 324)  o = softplusf(v);      // bw
    else if (c < 389)   o = sigf(v);           // e
    else if (c < 453)   o = v;                 // v
    else if (c < 459)   o = sigf(v);           // free, ga, gw
    else {                                     // Pi: softmax over groups of 3
        int h = (c - 459) / 3, base = 459 + h * 3;
        float a0 = row[base], a1 = row[base + 1], a2 = row[base + 2];
        float mx = fmaxf(a0, fmaxf(a1, a2));
        float e0 = expf(a0 - mx), e1 = expf(a1 - mx), e2 = expf(a2 - mx);
        o = expf(v - mx) / (e0 + e1 + e2);
    }
    xout[(size_t)n * 480 + c] = o;
}

// ---------- DNC memory update: one workgroup per batch item ----------
__global__ void dnc_k(const float* __restrict__ xi,
                      float* __restrict__ uG, float* __restrict__ wwG,
                      float* __restrict__ MG, float* __restrict__ pG, float* __restrict__ LG,
                      float* __restrict__ wrG, float* __restrict__ rG,
                      float* __restrict__ out_t) {
    int n = blockIdx.x, tid = threadIdx.x;
    __shared__ float s_wr[512], s_u[128], s_ww[128], s_p[128];
    __shared__ float s_free[4], s_br[4], s_Pi[12], s_kw[64], s_e[64], s_v[64], s_Kr[256];
    __shared__ float s_a[128], s_cw[128], s_phi[128];
    __shared__ float s_b[512], s_f[512], s_cr[512];
    __shared__ int   s_rank[128];
    __shared__ float s_ga, s_gw, s_bw, s_kwn, s_wsum, s_Krn[4];

    const float* xn = xi + (size_t)n * 480;
    if (tid < 256) s_Kr[tid] = xn[tid];
    if (tid < 4)  s_br[tid] = xn[256 + tid];
    if (tid < 64) { s_kw[tid] = xn[260 + tid]; s_e[tid] = xn[325 + tid]; s_v[tid] = xn[389 + tid]; }
    if (tid < 4)  s_free[tid] = xn[453 + tid];
    if (tid == 0) { s_bw = xn[324]; s_ga = xn[457]; s_gw = xn[458]; }
    if (tid < 12) s_Pi[tid] = xn[459 + tid];
    for (int i = tid; i < 512; i += 256) s_wr[i] = wrG[(size_t)n * 512 + i];
    if (tid < 128) {
        s_u[tid]  = uG[n * 128 + tid];
        s_ww[tid] = wwG[n * 128 + tid];
        s_p[tid]  = pG[n * 128 + tid];
    }
    __syncthreads();
    float* Mn = MG + (size_t)n * 8192;    // 128 x 64
    float* Ln = LG + (size_t)n * 16384;   // 128 x 128

    if (tid < 128) {
        float ps = 1.f;
        for (int h = 0; h < 4; ++h) ps *= (1.f - s_free[h] * s_wr[h * 128 + tid]);
        s_u[tid] = (s_u[tid] + (1.f - s_u[tid]) * s_ww[tid]) * ps;
    }
    __syncthreads();
    if (tid < 128) {
        float uu = s_u[tid]; int rk = 0;
        for (int t2 = 0; t2 < 128; ++t2) {
            float ut = s_u[t2];
            rk += (ut < uu) || (ut == uu && t2 < tid);
        }
        s_rank[tid] = rk;
        s_phi[rk] = uu;
    }
    __syncthreads();
    if (tid == 0) {   // exclusive cumprod of sorted usage
        float prod = 1.f;
        for (int j = 0; j < 128; ++j) { float v0 = s_phi[j]; s_phi[j] = prod; prod *= v0; }
    }
    __syncthreads();
    if (tid < 128) s_a[tid] = (1.f - s_u[tid]) * s_phi[s_rank[tid]];
    if (tid == 0) {
        float s2 = 0.f;
        for (int d = 0; d < 64; ++d) s2 += s_kw[d] * s_kw[d];
        s_kwn = sqrtf(s2);
    }
    __syncthreads();
    if (tid < 128) {            // content write weights from OLD M
        const float* Mr = Mn + tid * 64;
        float dot = 0.f, nm = 0.f;
        for (int d = 0; d < 64; ++d) { float m = Mr[d]; dot += m * s_kw[d]; nm += m * m; }
        s_cw[tid] = s_bw * (dot / fmaxf(sqrtf(nm) * s_kwn, 1e-8f));
    }
    __syncthreads();
    if (tid == 0) {
        float mx = -1e30f;
        for (int s = 0; s < 128; ++s) mx = fmaxf(mx, s_cw[s]);
        float sum = 0.f;
        for (int s = 0; s < 128; ++s) { float e = expf(s_cw[s] - mx); s_cw[s] = e; sum += e; }
        float inv = 1.f / sum;
        for (int s = 0; s < 128; ++s) s_cw[s] *= inv;
    }
    __syncthreads();
    if (tid < 128) s_ww[tid] = s_gw * (s_ga * s_a[tid] + (1.f - s_ga) * s_cw[tid]);
    __syncthreads();
    for (int i = tid; i < 8192; i += 256) {          // memory write
        int s = i >> 6, d = i & 63;
        Mn[i] = Mn[i] * (1.f - s_ww[s] * s_e[d]) + s_ww[s] * s_v[d];
    }
    for (int i = tid; i < 16384; i += 256) {         // link matrix (old p)
        int s = i >> 7, t2 = i & 127;
        float l = (1.f - s_ww[s] - s_ww[t2]) * Ln[i] + s_ww[t2] * s_p[s];
        Ln[i] = (s == t2) ? 0.f : l;
    }
    __syncthreads();
    if (tid == 0) {
        float ss = 0.f;
        for (int s = 0; s < 128; ++s) ss += s_ww[s];
        s_wsum = ss;
    }
    __syncthreads();
    if (tid < 128) s_p[tid] = (1.f - s_wsum) * s_p[tid] + s_ww[tid];
    if (tid < 4) {
        float s2 = 0.f;
        for (int d = 0; d < 64; ++d) { float k = s_Kr[tid * 64 + d]; s2 += k * k; }
        s_Krn[tid] = sqrtf(s2);
    }
    __syncthreads();
    for (int i = tid; i < 512; i += 256) {           // b/f from NEW L, OLD wr
        int h = i >> 7, t2 = i & 127;
        const float* wrh = s_wr + h * 128;
        float bb = 0.f, ff = 0.f;
        for (int s = 0; s < 128; ++s) {
            bb += wrh[s] * Ln[s * 128 + t2];
            ff += wrh[s] * Ln[t2 * 128 + s];
        }
        s_b[i] = bb; s_f[i] = ff;
    }
    for (int i = tid; i < 512; i += 256) {           // cr from NEW M
        int h = i >> 7, s = i & 127;
        const float* Mr = Mn + s * 64;
        const float* kr = s_Kr + h * 64;
        float dot = 0.f, nm = 0.f;
        for (int d = 0; d < 64; ++d) { float m = Mr[d]; dot += m * kr[d]; nm += m * m; }
        s_cr[i] = s_br[h] * (dot / fmaxf(sqrtf(nm) * s_Krn[h], 1e-8f));
    }
    __syncthreads();
    if (tid < 4) {
        float* row = s_cr + tid * 128;
        float mx = -1e30f;
        for (int s = 0; s < 128; ++s) mx = fmaxf(mx, row[s]);
        float sum = 0.f;
        for (int s = 0; s < 128; ++s) { float e = expf(row[s] - mx); row[s] = e; sum += e; }
        float inv = 1.f / sum;
        for (int s = 0; s < 128; ++s) row[s] *= inv;
    }
    __syncthreads();
    for (int i = tid; i < 512; i += 256) {
        int h = i >> 7;
        s_wr[i] = s_Pi[h * 3] * s_b[i] + s_Pi[h * 3 + 1] * s_cr[i] + s_Pi[h * 3 + 2] * s_f[i];
    }
    __syncthreads();
    float* on = out_t + (size_t)n * ROWSZ;
    if (tid < 256) {                                 // r = wr @ Mnew
        int h = tid >> 6, d = tid & 63;
        float acc = 0.f;
        for (int s = 0; s < 128; ++s) acc += s_wr[h * 128 + s] * Mn[s * 64 + d];
        rG[n * 256 + tid] = acc;
        on[OFF_R + tid] = acc;
    }
    if (tid < 128) {
        uG[n * 128 + tid] = s_u[tid];   on[OFF_U + tid] = s_u[tid];
        wwG[n * 128 + tid] = s_ww[tid]; on[OFF_WW + tid] = s_ww[tid];
        pG[n * 128 + tid] = s_p[tid];   on[OFF_P + tid] = s_p[tid];
    }
    for (int i = tid; i < 512; i += 256) { wrG[(size_t)n * 512 + i] = s_wr[i]; on[OFF_WR + i] = s_wr[i]; }
    for (int i = tid; i < 8192; i += 256)  on[OFF_M + i] = Mn[i];
    for (int i = tid; i < 16384; i += 256) on[OFF_L + i] = Ln[i];
}

// ---------- xa = [hT | r] in f16 ----------
__global__ void build_xa_k(const float* __restrict__ hT, const float* __restrict__ r,
                           _Float16* __restrict__ xah) {
    int i = blockIdx.x * blockDim.x + threadIdx.x;     // 128*768
    if (i >= 128 * 768) return;
    int n = i / 768, c = i % 768;
    float v = (c < 512) ? hT[n * 512 + c] : r[n * 256 + (c - 512)];
    xah[i] = (_Float16)v;
}

// ---------- actor softmax + critic value ----------
__global__ void actor_out_k(const float* __restrict__ logits, const float* __restrict__ hT,
                            const float* __restrict__ r,
                            const float* __restrict__ wcrit, const float* __restrict__ bcrit,
                            float* __restrict__ out_t) {
    int n = blockIdx.x, tid = threadIdx.x;
    __shared__ float red[256];
    float acc = 0.f;
    for (int c = tid; c < 768; c += 256) {
        float xv = (c < 512) ? hT[(size_t)n * 512 + c] : r[(size_t)n * 256 + (c - 512)];
        acc += xv * wcrit[c];
    }
    red[tid] = acc;
    __syncthreads();
    for (int s = 128; s > 0; s >>= 1) { if (tid < s) red[tid] += red[tid + s]; __syncthreads(); }
    float* on = out_t + (size_t)n * ROWSZ;
    if (tid == 0) {
        on[OFF_V] = red[0] + bcrit[0];
        const float* lg = logits + n * 16;
        float mx = -1e30f;
        for (int j = 0; j < 16; ++j) mx = fmaxf(mx, lg[j]);
        float sum = 0.f, e[16];
        for (int j = 0; j < 16; ++j) { e[j] = expf(lg[j] - mx); sum += e[j]; }
        float inv = 1.f / sum;
        for (int j = 0; j < 16; ++j) on[OFF_PROBS + j] = e[j] * inv;
    }
}

extern "C" void kernel_launch(void* const* d_in, const int* in_sizes, int n_in,
                              void* d_out, int out_size, void* d_ws, size_t ws_size,
                              hipStream_t stream) {
    (void)in_sizes; (void)n_in; (void)out_size; (void)ws_size;
    const float* inputs   = (const float*)d_in[0];
    const float* rnn_hxs  = (const float*)d_in[1];
    const float* w_ih0    = (const float*)d_in[2];
    const float* w_hh0    = (const float*)d_in[3];
    const float* b_ih0    = (const float*)d_in[4];
    const float* b_hh0    = (const float*)d_in[5];
    const float* w_ih1    = (const float*)d_in[6];
    const float* w_hh1    = (const float*)d_in[7];
    const float* b_ih1    = (const float*)d_in[8];
    const float* b_hh1    = (const float*)d_in[9];
    const float* w_xi     = (const float*)d_in[10];
    const float* b_xi     = (const float*)d_in[11];
    const float* w_actor  = (const float*)d_in[12];
    const float* b_actor  = (const float*)d_in[13];
    const float* w_critic = (const float*)d_in[14];
    const float* b_critic = (const float*)d_in[15];
    float* out = (float*)d_out;

    char* base = (char*)d_ws;
    size_t off = 0;
    auto alloc = [&](size_t bytes) -> void* {
        void* p = base + off;
        off += (bytes + 255) & ~(size_t)255;
        return p;
    };
    // f16 weights (wxi padded to 480 rows)
    _Float16* wih0h = (_Float16*)alloc(768 * 384 * 2);
    _Float16* whh0h = (_Float16*)alloc(768 * 256 * 2);
    _Float16* wih1h = (_Float16*)alloc(768 * 256 * 2);
    _Float16* whh1h = (_Float16*)alloc(768 * 256 * 2);
    _Float16* wxih  = (_Float16*)alloc(480 * 512 * 2);
    _Float16* wacth = (_Float16*)alloc(16 * 768 * 2);
    float*    bxip  = (float*)alloc(480 * 4);          // padded b_xi
    // f16 activation buffers (GEMM A operands)
    _Float16* xh    = (_Float16*)alloc(128 * 384 * 2);
    _Float16* h0h   = (_Float16*)alloc(128 * 256 * 2);
    _Float16* h1h   = (_Float16*)alloc(128 * 256 * 2);
    _Float16* h0nh  = (_Float16*)alloc(128 * 256 * 2);
    _Float16* h1nh  = (_Float16*)alloc(128 * 256 * 2);
    _Float16* hTrh  = (_Float16*)alloc(128 * 512 * 2); // relu(hT) f16
    _Float16* xah   = (_Float16*)alloc(128 * 768 * 2);
    // f32 buffers
    float* gi     = (float*)alloc(128 * 768 * 4);
    float* gh     = (float*)alloc(128 * 768 * 4);
    float* h0     = (float*)alloc(128 * 256 * 4);
    float* h1     = (float*)alloc(128 * 256 * 4);
    float* h0n    = (float*)alloc(128 * 256 * 4);
    float* h1n    = (float*)alloc(128 * 256 * 4);
    float* hT     = (float*)alloc(128 * 512 * 4);
    float* xiraw  = (float*)alloc(128 * 480 * 4);
    float* xipost = (float*)alloc(128 * 480 * 4);
    float* logits = (float*)alloc(128 * 16 * 4);
    float* rS     = (float*)alloc(128 * 256 * 4);
    float* wrS    = (float*)alloc(128 * 512 * 4);
    float* uS     = (float*)alloc(128 * 128 * 4);
    float* wwS    = (float*)alloc(128 * 128 * 4);
    float* pS     = (float*)alloc(128 * 128 * 4);
    float* MS     = (float*)alloc((size_t)128 * 8192 * 4);
    float* LS     = (float*)alloc((size_t)128 * 16384 * 4);

    auto cvt = [&](const float* s, _Float16* d, int nvalid, int ntotal) {
        cvt_pad_f16_k<<<(ntotal + 255) / 256, 256, 0, stream>>>(s, d, nvalid, ntotal);
    };
    cvt(w_ih0, wih0h, 768 * 384, 768 * 384);
    cvt(w_hh0, whh0h, 768 * 256, 768 * 256);
    cvt(w_ih1, wih1h, 768 * 256, 768 * 256);
    cvt(w_hh1, whh1h, 768 * 256, 768 * 256);
    cvt(w_xi,  wxih,  471 * 512, 480 * 512);
    cvt(w_actor, wacth, 16 * 768, 16 * 768);
    pad_bias_k<<<2, 256, 0, stream>>>(b_xi, bxip, 471, 480);
    init_state_k<<<(128 * 26240 + 255) / 256, 256, 0, stream>>>(
        rnn_hxs, h0, h1, h0h, h1h, rS, wrS, uS, wwS, MS, pS, LS);

    auto gemm = [&](const _Float16* A, const _Float16* W, const float* b, float* C,
                    int M_, int K_, int Nc_, int ldc_) {
        int waves = (M_ >> 4) * (Nc_ >> 4);
        int blocks = (waves + 7) >> 3;   // 8 waves (256 threads) per block
        switch (K_) {
            case 256: wmma_gemm_k<256><<<blocks, 256, 0, stream>>>(A, W, b, C, M_, Nc_, ldc_); break;
            case 384: wmma_gemm_k<384><<<blocks, 256, 0, stream>>>(A, W, b, C, M_, Nc_, ldc_); break;
            case 512: wmma_gemm_k<512><<<blocks, 256, 0, stream>>>(A, W, b, C, M_, Nc_, ldc_); break;
            default:  wmma_gemm_k<768><<<blocks, 256, 0, stream>>>(A, W, b, C, M_, Nc_, ldc_); break;
        }
    };

    for (int t = 0; t < 32; ++t) {
        const float* in_t = inputs + (size_t)t * 128 * 129;
        float* out_t = out + (size_t)t * 128 * ROWSZ;
        build_x_k<<<192, 256, 0, stream>>>(in_t, rS, xh);
        gemm(xh,  wih0h, b_ih0, gi, 128, 384, 768, 768);
        gemm(h0h, whh0h, b_hh0, gh, 128, 256, 768, 768);
        gru_combine_k<<<128, 256, 0, stream>>>(gi, gh, h0, h0n, h0nh);
        gemm(h0nh, wih1h, b_ih1, gi, 128, 256, 768, 768);
        gemm(h1h,  whh1h, b_hh1, gh, 128, 256, 768, 768);
        gru_combine_k<<<128, 256, 0, stream>>>(gi, gh, h1, h1n, h1nh);
        commit_h_k<<<256, 256, 0, stream>>>(h0n, h1n, h0, h1, h0h, h1h, hT, hTrh, out_t, in_t);
        gemm(hTrh, wxih, bxip, xiraw, 128, 512, 480, 480);
        xi_act_k<<<236, 256, 0, stream>>>(xiraw, xipost);
        dnc_k<<<128, 256, 0, stream>>>(xipost, uS, wwS, MS, pS, LS, wrS, rS, out_t);
        build_xa_k<<<384, 256, 0, stream>>>(hT, rS, xah);
        gemm(xah, wacth, b_actor, logits, 128, 768, 16, 16);
        actor_out_k<<<128, 256, 0, stream>>>(logits, hT, rS, w_critic, b_critic, out_t);
    }
}
